// CausalSelfAttention_3822520893703
// MI455X (gfx1250) — compile-verified, matrix operand order checked by reading, not run on hardware
//
#include <hip/hip_runtime.h>

// ---------------- types / constants ----------------
typedef __attribute__((ext_vector_type(16))) __bf16 v16bf;
typedef __attribute__((ext_vector_type(8)))  __bf16 v8bf;
typedef __attribute__((ext_vector_type(8)))  float  v8f;
typedef unsigned short     u16;
typedef unsigned int       u32;
typedef unsigned long long u64;
typedef __attribute__((ext_vector_type(4))) u32 su4;
typedef __attribute__((ext_vector_type(8))) u32 su8;

constexpr int Bb  = 4;
constexpr int Tt  = 2048;
constexpr int Cc  = 1024;
constexpr int Hh  = 16;
constexpr int Dd  = 64;
constexpr int NQT = Tt / 16;   // 128 query tiles per (b,h)

#define USE_TDM 1

// ---------------- helpers ----------------
static __device__ __forceinline__ u16 f2bf(float f) {
  u32 u = __builtin_bit_cast(u32, f);
  u += 0x7FFFu + ((u >> 16) & 1u);          // round-to-nearest-even
  return (u16)(u >> 16);
}
static __device__ __forceinline__ __bf16 tob(float f) {
  return __builtin_bit_cast(__bf16, f2bf(f));
}
static __device__ __forceinline__ v16bf cat8(v8bf lo, v8bf hi) {
  v16bf r;
#pragma unroll
  for (int i = 0; i < 8; ++i) { r[i] = lo[i]; r[i + 8] = hi[i]; }
  return r;
}
// 16-bit WMMA fragment (16 elems/lane): K groups {hf*8..+7} and {16+hf*8..+7}
static __device__ __forceinline__ v16bf ldfrag(const u16* p, int hf) {
  v8bf lo = *(const v8bf*)(p + hf * 8);
  v8bf hi = *(const v8bf*)(p + 16 + hf * 8);
  return cat8(lo, hi);
}

#if USE_TDM
// Tensor Data Mover: DMA a 64x32 bf16 tile (row stride K elems) from global
// into LDS at lds_off, inserting 16B padding after every 64B row so the LDS
// layout is the same stride-40 (u16) layout the fragment loaders expect.
// D# bitfields per CDNA5 ISA (group0 128b, group1 256b).
static __device__ __forceinline__ void tdm_load_a_tile(
    u32 lds_off, const u16* gsrc, u32 K, u32 M)
{
  u64 ga = (u64)(uintptr_t)gsrc;
  su4 g0;
  g0[0] = 1u;                                            // count=1, user D#
  g0[1] = lds_off;                                       // lds_addr (bytes)
  g0[2] = (u32)ga;                                       // global_addr[31:0]
  g0[3] = ((u32)(ga >> 32) & 0x01FFFFFFu) | (2u << 30);  // addr[56:32] | type=2
  su8 g1;
  g1[0] = (1u << 16)                                     // data_size = 2 bytes
        | (1u << 20)                                     // pad_enable
        | (3u << 22)                                     // pad_interval: 16 DW (one row)
        | (3u << 25);                                    // pad_amount: 4 DW (16B)
  g1[1] = (K & 0xFFFFu) << 16;                           // tensor_dim0[15:0]
  g1[2] = (K >> 16) | ((M & 0xFFFFu) << 16);             // td0[31:16] | td1[15:0]
  g1[3] = (M >> 16) | (32u << 16);                       // td1[31:16] | tile_dim0=32
  g1[4] = 64u;                                           // tile_dim1=64, tile_dim2=0
  g1[5] = K;                                             // tensor_dim0_stride[31:0]
  g1[6] = 0u;
  g1[7] = 0u;
  asm volatile("tensor_load_to_lds %0, %1" :: "s"(g0), "s"(g1) : "memory");
}
#endif

// ---------------- f32 -> bf16 convert ----------------
__global__ void cvt_kernel(const float* __restrict__ src, u16* __restrict__ dst, int n) {
  for (int i = blockIdx.x * blockDim.x + threadIdx.x; i < n; i += gridDim.x * blockDim.x)
    dst[i] = f2bf(src[i]);
}

// ---------------- bf16 WMMA GEMM: C[M,N] = A[M,K] * B[K,N] + bias ----------------
// Block: 256 threads (8 waves), tile 64(M) x 128(N), K-step 32.
// A tile staged by the Tensor Data Mover (wave 0 issues, TENSORcnt-tracked),
// overlapped with the B-tile load+transpose done by all waves.
// EPI==0: QKV epilogue -> scatter bf16 q[b,h,t,d], k[b,h,t,d], vT[b,h,d,t]
// EPI==1: proj epilogue -> fp32 C
template <int EPI>
__global__ __launch_bounds__(256) void gemm_bf16(
    const u16* __restrict__ gA, const u16* __restrict__ gB,
    const float* __restrict__ bias, int M, int N, int K,
    u16* __restrict__ q_o, u16* __restrict__ k_o, u16* __restrict__ vt_o,
    float* __restrict__ c_o)
{
  __shared__ u16 Asm[64 * 40];    // A tile row-major, padded stride 40
  __shared__ u16 Bsm[128 * 40];   // B tile stored transposed [n][k], stride 40

  const int tid  = threadIdx.x;
  const int m0   = blockIdx.y * 64;
  const int n0   = blockIdx.x * 128;
  const int wid  = tid >> 5, lane = tid & 31;
  const int hf   = lane >> 4, mn = lane & 15;
  const int wm   = wid & 1,  wn = wid >> 1;       // 2 x 4 wave grid of 32x32 subtiles

  const int ar  = tid >> 2, ac  = (tid & 3) * 8;  // A fallback loader
  const int bkr = tid >> 3, bnc = (tid & 7) * 16; // B loader: 16 bf16 per thread

#if USE_TDM
  const u32 ldsA = (u32)(uintptr_t)(&Asm[0]);     // LDS aperture: addr[31:0] = offset
#endif

  v8f acc[2][2] = {};

  for (int k0 = 0; k0 < K; k0 += 32) {
#if USE_TDM
    if (wid == 0)                                  // one TDM op per workgroup
      tdm_load_a_tile(ldsA, gA + (size_t)m0 * K + k0, (u32)K, (u32)M);
#else
    uint4 av = *(const uint4*)(gA + (size_t)(m0 + ar) * K + k0 + ac);
    *(uint4*)&Asm[ar * 40 + ac] = av;
#endif
    const u16* bp = gB + (size_t)(k0 + bkr) * N + n0 + bnc;
    uint4 b0 = *(const uint4*)bp;
    uint4 b1 = *(const uint4*)(bp + 8);
    if (k0 + 32 < K)  // -> global_prefetch_b8
      __builtin_prefetch(gB + (size_t)(k0 + 32 + bkr) * N + n0 + bnc, 0, 1);
    alignas(16) u16 tmp[16];
    *(uint4*)tmp       = b0;
    *(uint4*)(tmp + 8) = b1;
#pragma unroll
    for (int j = 0; j < 16; ++j) Bsm[(bnc + j) * 40 + bkr] = tmp[j];
#if USE_TDM
    if (wid == 0)
      __builtin_amdgcn_s_wait_tensorcnt(0);        // s_wait_tensorcnt 0x0
#endif
    __syncthreads();

    v16bf af[2], bfv[2];
#pragma unroll
    for (int a = 0; a < 2; ++a)
      af[a] = ldfrag(&Asm[(wm * 32 + a * 16 + mn) * 40], hf);
#pragma unroll
    for (int b2 = 0; b2 < 2; ++b2)
      bfv[b2] = ldfrag(&Bsm[(wn * 32 + b2 * 16 + mn) * 40], hf);
#pragma unroll
    for (int a = 0; a < 2; ++a)
#pragma unroll
      for (int b2 = 0; b2 < 2; ++b2)
        acc[a][b2] = __builtin_amdgcn_wmma_f32_16x16x32_bf16(
            false, af[a], false, bfv[b2], (short)0, acc[a][b2], false, false);
    __syncthreads();
  }

#pragma unroll
  for (int a = 0; a < 2; ++a) {
#pragma unroll
    for (int b2 = 0; b2 < 2; ++b2) {
#pragma unroll
      for (int r = 0; r < 8; ++r) {
        int gm = m0 + wm * 32 + a * 16 + hf * 8 + r;   // C/D layout: row = r (+8 for hi half)
        int gn = n0 + wn * 32 + b2 * 16 + mn;          // col = lane
        float val = acc[a][b2][r] + bias[gn];
        if (EPI == 0) {
          int which = gn >> 10, cc2 = gn & 1023;       // 0=q 1=k 2=v
          int hh = cc2 >> 6, dd = cc2 & 63;
          int bi = gm >> 11, tt2 = gm & 2047;
          int bh2 = bi * Hh + hh;
          u16 bv = f2bf(val);
          if (which == 0)      q_o[((size_t)bh2 * Tt + tt2) * Dd + dd] = bv;
          else if (which == 1) k_o[((size_t)bh2 * Tt + tt2) * Dd + dd] = bv;
          else                 vt_o[((size_t)bh2 * Dd + dd) * Tt + tt2] = bv;  // transposed
        } else {
          c_o[(size_t)gm * N + gn] = val;
        }
      }
    }
  }
}

// ---------------- fused causal attention ----------------
// One workgroup (8 waves) per (b, h, 16-row query tile).
// Full 16 x T fp32 score strip lives in LDS (128KB of the 320KB/WGP).
__global__ __launch_bounds__(256) void attn_kernel(
    const u16* __restrict__ qb, const u16* __restrict__ kb,
    const u16* __restrict__ vtb, float* __restrict__ att,
    u16* __restrict__ yb)
{
  extern __shared__ char smem[];
  float* s_strip = (float*)smem;                                 // 16*T f32
  u16*   q_lds   = (u16*)  (smem + 16 * Tt * 4);                 // 16*64 bf16
  float* y_lds   = (float*)(smem + 16 * Tt * 4 + 2048);          // 16*64 f32
  float* red     = (float*)(smem + 16 * Tt * 4 + 2048 + 4096);   // 16*16 f32
  float* rowm    = red + 256;                                    // 16 f32
  float* rowl    = rowm + 16;                                    // 16 f32 (stores 1/l)

  const int tid = threadIdx.x;
  const int bh  = blockIdx.x / NQT;
  const int qt  = blockIdx.x % NQT;
  const int b   = bh >> 4, h = bh & 15;
  const int wid = tid >> 5, lane = tid & 31;
  const int hf  = lane >> 4, mn = lane & 15;
  const float NEGINF = -__builtin_inff();

  { // stage q tile to LDS, zero y accumulator
    int idx4 = tid * 4, row = idx4 >> 6, col = idx4 & 63;
    const u16* qsrc = qb + ((size_t)bh * Tt + qt * 16 + row) * Dd + col;
    *(uint2*)&q_lds[row * 64 + col] = *(const uint2*)qsrc;
    float4 z4 = make_float4(0.f, 0.f, 0.f, 0.f);
    *(float4*)&y_lds[idx4] = z4;
  }
  __syncthreads();

  v16bf aq[2];
#pragma unroll
  for (int c = 0; c < 2; ++c) aq[c] = ldfrag(q_lds + mn * 64 + c * 32, hf);

  // ---- S = (q . k^T)/8 over causal key tiles, striped across 8 waves ----
  for (int kt = wid; kt <= qt; kt += 8) {
    const u16* kp = kb + ((size_t)bh * Tt + kt * 16 + mn) * Dd;  // lane = key column
    v16bf bk0 = ldfrag(kp, hf);
    v16bf bk1 = ldfrag(kp + 32, hf);
    v8f s = {};
    s = __builtin_amdgcn_wmma_f32_16x16x32_bf16(false, aq[0], false, bk0, (short)0, s, false, false);
    s = __builtin_amdgcn_wmma_f32_16x16x32_bf16(false, aq[1], false, bk1, (short)0, s, false, false);
#pragma unroll
    for (int r = 0; r < 8; ++r) {
      int mrow = hf * 8 + r;
      float v = s[r] * 0.125f;
      if (kt == qt && mn > mrow) v = NEGINF;                     // causal mask inside diag tile
      s_strip[mrow * Tt + kt * 16 + mn] = v;
    }
  }
  __syncthreads();

  const int L  = (qt + 1) * 16;                                  // valid key length
  const int rl = tid >> 4, cl = tid & 15;

  // ---- row max ----
  float mx = NEGINF;
  for (int c = cl; c < L; c += 16) mx = fmaxf(mx, s_strip[rl * Tt + c]);
  red[rl * 16 + cl] = mx;
  __syncthreads();
  if (tid < 16) {
    float m = red[tid * 16];
#pragma unroll
    for (int j = 1; j < 16; ++j) m = fmaxf(m, red[tid * 16 + j]);
    rowm[tid] = m;
  }
  __syncthreads();

  // ---- exp + row sum (stores e in strip) ----
  float rm = rowm[rl], sacc = 0.f;
  for (int c = cl; c < L; c += 16) {
    float e = __expf(s_strip[rl * Tt + c] - rm);
    s_strip[rl * Tt + c] = e;
    sacc += e;
  }
  red[rl * 16 + cl] = sacc;
  __syncthreads();
  if (tid < 16) {
    float s = 0.f;
#pragma unroll
    for (int j = 0; j < 16; ++j) s += red[tid * 16 + j];
    rowl[tid] = 1.f / s;
  }
  __syncthreads();

  // ---- normalize, emit att, zero causal tail ----
  float* attb = att + ((size_t)bh * Tt + qt * 16) * (size_t)Tt;
  for (int i = tid; i < 16 * L; i += 256) {
    int r = i / L, c = i - r * L;
    float p = s_strip[r * Tt + c] * rowl[r];
    s_strip[r * Tt + c] = p;
    attb[(size_t)r * Tt + c] = p;
  }
  int L32 = (L + 31) & ~31;
  if (L32 > L) {                                                 // pad strip to 32-mult for WMMA
    int r = tid >> 4, c = L + (tid & 15);
    s_strip[r * Tt + c] = 0.f;
  }
  if (L < Tt) {                                                  // zeros above the diagonal
    int W4 = (Tt - L) >> 2, tot = 16 * W4;
    float4 z4 = make_float4(0.f, 0.f, 0.f, 0.f);
    for (int i = tid; i < tot; i += 256) {
      int r = i / W4, c4 = i - r * W4;
      *(float4*)(attb + (size_t)r * Tt + L + c4 * 4) = z4;
    }
  }
  __syncthreads();

  // ---- y = P @ V, K striped across 8 waves, reduced through LDS atomics ----
  v8f yacc[4] = {};
  int nCh = L32 >> 5;
  for (int ch = wid; ch < nCh; ch += 8) {
    const float* prow = s_strip + mn * Tt + ch * 32;             // lane = p row
    v16bf ap;
#pragma unroll
    for (int g = 0; g < 2; ++g) {
      float4 f0 = *(const float4*)(prow + g * 16 + hf * 8);
      float4 f1 = *(const float4*)(prow + g * 16 + hf * 8 + 4);
      ap[g*8+0]=tob(f0.x); ap[g*8+1]=tob(f0.y); ap[g*8+2]=tob(f0.z); ap[g*8+3]=tob(f0.w);
      ap[g*8+4]=tob(f1.x); ap[g*8+5]=tob(f1.y); ap[g*8+6]=tob(f1.z); ap[g*8+7]=tob(f1.w);
    }
#pragma unroll
    for (int dt = 0; dt < 4; ++dt) {
      const u16* vp = vtb + ((size_t)bh * Dd + dt * 16 + mn) * Tt + ch * 32;  // vT: contiguous K
      v16bf bv = ldfrag(vp, hf);
      yacc[dt] = __builtin_amdgcn_wmma_f32_16x16x32_bf16(
          false, ap, false, bv, (short)0, yacc[dt], false, false);
    }
  }
#pragma unroll
  for (int dt = 0; dt < 4; ++dt)
#pragma unroll
    for (int r = 0; r < 8; ++r)
      atomicAdd(&y_lds[(hf * 8 + r) * 64 + dt * 16 + mn], yacc[dt][r]);   // ds_add_f32
  __syncthreads();

  { // y tile -> bf16 [B,T,C] for proj GEMM
    int idx4 = tid * 4, row = idx4 >> 6, col = idx4 & 63;
    u16* dst = yb + ((size_t)(b * Tt + qt * 16 + row)) * Cc + h * 64 + col;
    u32 p0 = (u32)f2bf(y_lds[idx4 + 0]) | ((u32)f2bf(y_lds[idx4 + 1]) << 16);
    u32 p1 = (u32)f2bf(y_lds[idx4 + 2]) | ((u32)f2bf(y_lds[idx4 + 3]) << 16);
    uint2 pk; pk.x = p0; pk.y = p1;
    *(uint2*)dst = pk;
  }
}

// ---------------- launcher ----------------
extern "C" void kernel_launch(void* const* d_in, const int* in_sizes, int n_in,
                              void* d_out, int out_size, void* d_ws, size_t ws_size,
                              hipStream_t stream) {
  (void)in_sizes; (void)n_in; (void)out_size; (void)ws_size;
  const float* x     = (const float*)d_in[0];
  const float* Wqkv  = (const float*)d_in[1];
  const float* bqkv  = (const float*)d_in[2];
  const float* Wproj = (const float*)d_in[3];
  const float* bproj = (const float*)d_in[4];
  // d_in[5] (attn_mask) unused: causality is computed analytically.

  float* y_out   = (float*)d_out;
  float* att_out = y_out + (size_t)Bb * Tt * Cc;

  const size_t nX  = (size_t)Bb * Tt * Cc;   // 8388608
  const size_t nWq = (size_t)Cc * 3 * Cc;    // 3145728
  const size_t nWp = (size_t)Cc * Cc;        // 1048576

  char* ws = (char*)d_ws;
  u16* xb     = (u16*)ws; ws += nX  * 2;
  u16* wqkvb  = (u16*)ws; ws += nWq * 2;
  u16* wprojb = (u16*)ws; ws += nWp * 2;
  u16* qb     = (u16*)ws; ws += nX  * 2;
  u16* kb     = (u16*)ws; ws += nX  * 2;
  u16* vtb    = (u16*)ws; ws += nX  * 2;
  u16* yb     = (u16*)ws; ws += nX  * 2;    // ~92.3 MB total

  cvt_kernel<<<2048, 256, 0, stream>>>(x, xb, (int)nX);
  cvt_kernel<<<2048, 256, 0, stream>>>(Wqkv, wqkvb, (int)nWq);
  cvt_kernel<<<1024, 256, 0, stream>>>(Wproj, wprojb, (int)nWp);

  gemm_bf16<0><<<dim3(3 * Cc / 128, Bb * Tt / 64), 256, 0, stream>>>(
      xb, wqkvb, bqkv, Bb * Tt, 3 * Cc, Cc, qb, kb, vtb, nullptr);

  size_t attn_lds = (size_t)16 * Tt * 4 + 2048 + 4096 + 256 * 4 + 32 * 4;  // 138368 B
  attn_kernel<<<Bb * Hh * NQT, 256, attn_lds, stream>>>(qb, kb, vtb, att_out, yb);

  gemm_bf16<1><<<dim3(Cc / 128, Bb * Tt / 64), 256, 0, stream>>>(
      yb, wprojb, bproj, Bb * Tt, Cc, Cc, nullptr, nullptr, nullptr, y_out);
}
